// MHA_48593259987157
// MI455X (gfx1250) — compile-verified, matrix operand order checked by reading, not run on hardware
//
#include <hip/hip_runtime.h>

// ---------------------------------------------------------------------------
// MHA forward for MI455X (gfx1250, wave32, WMMA + async-to-LDS).
//   attn [B,H,S,S] f32 and output [B,S,D] f32 concatenated in d_out.
// Stages:
//   0) transpose+cvt weights f32[K][N] -> bf16[N][K] (once, 4 matrices)
//   1) Q/K/V proj GEMMs (A=f32 inputs, B=bf16 W^T) -> bf16 in ws
//   2) scores = Q@K^T * 0.125 per head -> f32 raw into d_out attn region
//   3) row softmax in place (vectorized, shfl reductions)
//   4) ctx = P@V per head (BN=64 config) -> bf16 O in ws
//   5) output = O@Wo^T + bo -> f32
// GEMM core: v_wmma_f32_16x16x32_bf16, double-buffered LDS, software
// pipeline. bf16 operand streams use GLOBAL_LOAD_ASYNC_TO_LDS_B128
// (ASYNCcnt); f32 streams pipeline through registers with cvt_pk_bf16.
// LDS rows padded to 80B for conflict-free b128 fragment loads.
// ---------------------------------------------------------------------------

typedef __attribute__((ext_vector_type(16))) __bf16 v16bf;
typedef __attribute__((ext_vector_type(8)))  __bf16 v8bf;
typedef __attribute__((ext_vector_type(8)))  float  v8f;

#define MHA_B   2
#define MHA_S   2048
#define MHA_D   1024
#define MHA_H   16
#define MHA_DH  64

#define BM  128
#define BK  32
#define BKP 40            // padded LDS row length in elems (80B, 16B-aligned)
#define NTHREADS 256

// ---- CDNA5 async global->LDS copy (16B per lane, tracked by ASYNCcnt) ------
__device__ __forceinline__ void async_load16(void* ldsDst, const void* gsrc) {
  // Low 32 bits of a flat shared address are the LDS byte offset (ISA 10.2).
  const unsigned       lds = (unsigned)(unsigned long long)ldsDst;
  const unsigned long long ga = (unsigned long long)gsrc;
  asm volatile("global_load_async_to_lds_b128 %0, %1, off"
               :: "v"(lds), "v"(ga) : "memory");
}
__device__ __forceinline__ void async_wait_all() {
  asm volatile("s_wait_asynccnt 0x0" ::: "memory");
}

__device__ __forceinline__ v8bf cvt8(const float4 a, const float4 b) {
  v8bf r;
  r[0] = (__bf16)a.x; r[1] = (__bf16)a.y; r[2] = (__bf16)a.z; r[3] = (__bf16)a.w;
  r[4] = (__bf16)b.x; r[5] = (__bf16)b.y; r[6] = (__bf16)b.z; r[7] = (__bf16)b.w;
  return r;
}

template<typename TO>
__device__ __forceinline__ void store_out(TO* p, float v);
template<> __device__ __forceinline__ void store_out<float>(float* p, float v)   { *p = v; }
template<> __device__ __forceinline__ void store_out<__bf16>(__bf16* p, float v) { *p = (__bf16)v; }

// Per ISA 7.12.2 (16-bit A/B 16x32 layout): lane L holds row/col = L&15,
// elements 0..7 <- K = half*8+0..7, elements 8..15 <- K = 16+half*8+0..7.
__device__ __forceinline__ v16bf load_frag(const __bf16* rowPtr, int half) {
  const v8bf lo = *(const v8bf*)(rowPtr + half * 8);
  const v8bf hi = *(const v8bf*)(rowPtr + 16 + half * 8);
  v16bf f;
#pragma unroll
  for (int i = 0; i < 8; ++i) { f[i] = lo[i]; f[8 + i] = hi[i]; }
  return f;
}

// ---------------------------------------------------------------------------
// Batched GEMM: C = scale*(A@B) + bias per (batch,head).  B is always bf16.
// BT=true : B given as [N][K] row-major -> async copy straight to LDS.
// BT=false: B given as [K][N] row-major -> register load + LDS scatter.
// A f32   : register pipeline (float4 loads, cvt_pk_bf16, ds_store_b128).
// A bf16  : async copy straight to LDS.
// BN_T=128: 8 waves as 4x2, each 32x64.  BN_T=64: 8 waves as 8x1, each 16x64.
// All dims exact multiples of the tile (guaranteed by launches).
// ---------------------------------------------------------------------------
template<typename TA, typename TO, bool BT, int BN_T>
__global__ __launch_bounds__(NTHREADS)
void gemm_wmma_bf16(const TA* __restrict__ A, int lda, long long aBat, long long aHead,
                    const __bf16* __restrict__ Bm, int ldb, long long bBat, long long bHead,
                    const float* __restrict__ bias,
                    TO* __restrict__ C, int ldc, long long cBat, long long cHead,
                    int K, int nHeads, float scale) {
  constexpr bool A_F32   = (sizeof(TA) == 4);
  constexpr bool USE_ASYNC = (!A_F32) || BT;
  constexpr int WN    = BN_T / 64;
  constexpr int WM    = 8 / WN;
  constexpr int WROWS = BM / WM;
  constexpr int TM    = WROWS / 16;
  constexpr int TN    = 4;
  constexpr int ACH   = (BM * BK / 8) / NTHREADS;     // A chunks per thread (2)
  constexpr int BCH   = (BN_T * BK / 8) / NTHREADS;   // B chunks per thread (2/1)
  constexpr int CPR   = BN_T / 8;                     // BT=false: chunks per k-row

  __shared__ __attribute__((aligned(16))) __bf16 sA[2][BM * BKP];
  __shared__ __attribute__((aligned(16))) __bf16 sB[2][BN_T * BKP];

  const int z  = blockIdx.z;
  const int bb = z / nHeads;
  const int hh = z % nHeads;
  A  += (long long)bb * aBat + (long long)hh * aHead;
  Bm += (long long)bb * bBat + (long long)hh * bHead;
  C  += (long long)bb * cBat + (long long)hh * cHead;

  const int blockM = blockIdx.y * BM;
  const int blockN = blockIdx.x * BN_T;

  const int tid   = threadIdx.x;
  const int wave  = tid >> 5;
  const int lane  = tid & 31;
  const int lrow  = lane & 15;
  const int half  = lane >> 4;
  const int waveM = wave / WN;
  const int waveN = wave % WN;

  // pipeline registers (only live in the instantiations that use them)
  float4 aReg[ACH][2];
  v8bf   bScat[BCH];

  v8f acc[TM][TN];
#pragma unroll
  for (int tm = 0; tm < TM; ++tm)
#pragma unroll
    for (int tn = 0; tn < TN; ++tn)
#pragma unroll
      for (int r = 0; r < 8; ++r) acc[tm][tn][r] = 0.0f;

  // ---- staging issue: start global traffic for tile at k0 into buf ----
  auto stage_issue = [&](int k0, int buf) {
#pragma unroll
    for (int i = 0; i < ACH; ++i) {
      const int li = i * NTHREADS + tid;
      const int r  = li >> 2;
      const int c8 = li & 3;
      const TA* src = A + (long long)(blockM + r) * lda + (k0 + c8 * 8);
      if constexpr (A_F32) {
        aReg[i][0] = *(const float4*)src;
        aReg[i][1] = *(const float4*)(src + 4);
      } else {
        async_load16(&sA[buf][r * BKP + c8 * 8], src);
      }
    }
    if constexpr (BT) {
#pragma unroll
      for (int i = 0; i < BCH; ++i) {
        const int li = i * NTHREADS + tid;
        const int n  = li >> 2;
        const int c8 = li & 3;
        async_load16(&sB[buf][n * BKP + c8 * 8],
                     Bm + (long long)(blockN + n) * ldb + (k0 + c8 * 8));
      }
    } else {
#pragma unroll
      for (int i = 0; i < BCH; ++i) {
        const int li = i * NTHREADS + tid;
        const int kk = li / CPR;
        const int n8 = li % CPR;
        bScat[i] = *(const v8bf*)(Bm + (long long)(k0 + kk) * ldb + (blockN + n8 * 8));
      }
    }
  };

  // ---- staging commit: move register payloads into LDS buf ----
  auto stage_commit = [&](int buf) {
    if constexpr (A_F32) {
#pragma unroll
      for (int i = 0; i < ACH; ++i) {
        const int li = i * NTHREADS + tid;
        const int r  = li >> 2;
        const int c8 = li & 3;
        *(v8bf*)&sA[buf][r * BKP + c8 * 8] = cvt8(aReg[i][0], aReg[i][1]);
      }
    }
    if constexpr (!BT) {
#pragma unroll
      for (int i = 0; i < BCH; ++i) {
        const int li = i * NTHREADS + tid;
        const int kk = li / CPR;
        const int n8 = li % CPR;
#pragma unroll
        for (int j = 0; j < 8; ++j) sB[buf][(n8 * 8 + j) * BKP + kk] = bScat[i][j];
      }
    }
    if constexpr (USE_ASYNC) async_wait_all();
  };

  // ---- software pipeline over k ----
  stage_issue(0, 0);
  stage_commit(0);
  __syncthreads();

  int buf = 0;
  for (int k0 = 0; k0 < K; k0 += BK) {
    const int  nb   = buf ^ 1;
    const bool more = (k0 + BK) < K;
    if (more) stage_issue(k0 + BK, nb);   // global traffic overlaps WMMA below

    v16bf afr[TM], bfr[TN];
#pragma unroll
    for (int tm = 0; tm < TM; ++tm)
      afr[tm] = load_frag(&sA[buf][(waveM * WROWS + tm * 16 + lrow) * BKP], half);
#pragma unroll
    for (int tn = 0; tn < TN; ++tn)
      bfr[tn] = load_frag(&sB[buf][(waveN * 64 + tn * 16 + lrow) * BKP], half);
#pragma unroll
    for (int tm = 0; tm < TM; ++tm)
#pragma unroll
      for (int tn = 0; tn < TN; ++tn)
        acc[tm][tn] = __builtin_amdgcn_wmma_f32_16x16x32_bf16(
            false, afr[tm], false, bfr[tn], (short)0, acc[tm][tn], false, false);

    if (more) stage_commit(nb);
    __syncthreads();
    buf = nb;
  }

  // ---- epilogue (C layout: VGPR r -> M=r (lanes 0-15) / r+8 (16-31)) ----
  float bcol[TN];
#pragma unroll
  for (int tn = 0; tn < TN; ++tn)
    bcol[tn] = bias ? bias[blockN + waveN * 64 + tn * 16 + lrow] : 0.0f;
#pragma unroll
  for (int tm = 0; tm < TM; ++tm)
#pragma unroll
    for (int tn = 0; tn < TN; ++tn)
#pragma unroll
      for (int r = 0; r < 8; ++r) {
        const int gr = blockM + waveM * WROWS + tm * 16 + r + half * 8;
        const int gc = blockN + waveN * 64 + tn * 16 + lrow;
        store_out<TO>(&C[(long long)gr * ldc + gc], acc[tm][tn][r] * scale + bcol[tn]);
      }
}

// ---------------------------------------------------------------------------
// Transpose + f32->bf16 convert a 1024x1024 weight: WT[n][k] = (bf16)W[k][n].
// ---------------------------------------------------------------------------
__global__ __launch_bounds__(NTHREADS)
void transpose_cvt_1024(const float* __restrict__ W, __bf16* __restrict__ WT) {
  __shared__ __attribute__((aligned(16))) __bf16 t[64 * 80];
  const int c0 = blockIdx.x * 64;
  const int r0 = blockIdx.y * 64;
  const int tid = threadIdx.x;
#pragma unroll
  for (int i = 0; i < 16; ++i) {             // 4096 elems, coalesced f32 reads
    const int li = i * NTHREADS + tid;
    const int r = li >> 6, c = li & 63;
    t[c * 80 + r] = (__bf16)W[(long long)(r0 + r) * MHA_D + c0 + c];
  }
  __syncthreads();
#pragma unroll
  for (int i = 0; i < 2; ++i) {              // 512 v8bf chunks, coalesced b128
    const int li = i * NTHREADS + tid;
    const int n = li >> 3, k8 = li & 7;
    *(v8bf*)&WT[(long long)(c0 + n) * MHA_D + r0 + k8 * 8] = *(const v8bf*)&t[n * 80 + k8 * 8];
  }
}

// ---------------------------------------------------------------------------
// In-place softmax over rows of 2048: float4 loads, shfl + 8-entry LDS reduce.
// ---------------------------------------------------------------------------
__global__ __launch_bounds__(NTHREADS)
void softmax_rows_2048(float* __restrict__ attn) {
  __shared__ float red[8];
  float* p = attn + (long long)blockIdx.x * MHA_S;
  const int t = threadIdx.x;
  const int lane = t & 31, wv = t >> 5;

  const float4 a = *(const float4*)(p + t * 8);
  const float4 b = *(const float4*)(p + t * 8 + 4);
  float v[8] = {a.x, a.y, a.z, a.w, b.x, b.y, b.z, b.w};

  float m = v[0];
#pragma unroll
  for (int i = 1; i < 8; ++i) m = fmaxf(m, v[i]);
#pragma unroll
  for (int off = 16; off > 0; off >>= 1) m = fmaxf(m, __shfl_xor(m, off, 32));
  if (lane == 0) red[wv] = m;
  __syncthreads();
  m = red[0];
#pragma unroll
  for (int w = 1; w < 8; ++w) m = fmaxf(m, red[w]);

  float s = 0.0f;
#pragma unroll
  for (int i = 0; i < 8; ++i) { v[i] = __expf(v[i] - m); s += v[i]; }
#pragma unroll
  for (int off = 16; off > 0; off >>= 1) s += __shfl_xor(s, off, 32);
  __syncthreads();
  if (lane == 0) red[wv] = s;
  __syncthreads();
  s = red[0];
#pragma unroll
  for (int w = 1; w < 8; ++w) s += red[w];
  const float inv = 1.0f / s;

  float4 oa = {v[0] * inv, v[1] * inv, v[2] * inv, v[3] * inv};
  float4 ob = {v[4] * inv, v[5] * inv, v[6] * inv, v[7] * inv};
  *(float4*)(p + t * 8) = oa;
  *(float4*)(p + t * 8 + 4) = ob;
}

extern "C" void kernel_launch(void* const* d_in, const int* in_sizes, int n_in,
                              void* d_out, int out_size, void* d_ws, size_t ws_size,
                              hipStream_t stream) {
  (void)in_sizes; (void)n_in; (void)out_size; (void)ws_size;

  const float* q  = (const float*)d_in[0];
  const float* k  = (const float*)d_in[1];
  const float* v  = (const float*)d_in[2];
  const float* Wq = (const float*)d_in[3];
  const float* bq = (const float*)d_in[4];
  const float* Wk = (const float*)d_in[5];
  const float* bk = (const float*)d_in[6];
  const float* Wv = (const float*)d_in[7];
  const float* bv = (const float*)d_in[8];
  const float* Wo = (const float*)d_in[9];
  const float* bo = (const float*)d_in[10];

  const long long M  = (long long)MHA_B * MHA_S;         // 4096
  const long long SD = (long long)MHA_S * MHA_D;
  const long long SS = (long long)MHA_S * MHA_S;
  const long long DD = (long long)MHA_D * MHA_D;
  const long long attnElems = (long long)MHA_B * MHA_H * SS;

  __bf16* Qb  = (__bf16*)d_ws;          // activations, bf16
  __bf16* Kb  = Qb + M * MHA_D;
  __bf16* Vb  = Kb + M * MHA_D;
  __bf16* Ob  = Vb + M * MHA_D;
  __bf16* WqT = Ob + M * MHA_D;         // transposed bf16 weights
  __bf16* WkT = WqT + DD;
  __bf16* WvT = WkT + DD;
  __bf16* WoT = WvT + DD;

  float* attn = (float*)d_out;
  float* outp = attn + attnElems;

  dim3 blk(NTHREADS);

  // 0) weight transpose + convert
  dim3 gT(MHA_D / 64, MHA_D / 64);
  transpose_cvt_1024<<<gT, blk, 0, stream>>>(Wq, WqT);
  transpose_cvt_1024<<<gT, blk, 0, stream>>>(Wk, WkT);
  transpose_cvt_1024<<<gT, blk, 0, stream>>>(Wv, WvT);
  transpose_cvt_1024<<<gT, blk, 0, stream>>>(Wo, WoT);

  // 1) Q/K/V projections
  dim3 gProj(MHA_D / 128, (unsigned)(M / BM), 1);        // 8 x 32
  gemm_wmma_bf16<float, __bf16, true, 128><<<gProj, blk, 0, stream>>>(
      q, MHA_D, 0, 0, WqT, MHA_D, 0, 0, bq, Qb, MHA_D, 0, 0, MHA_D, 1, 1.0f);
  gemm_wmma_bf16<float, __bf16, true, 128><<<gProj, blk, 0, stream>>>(
      k, MHA_D, 0, 0, WkT, MHA_D, 0, 0, bk, Kb, MHA_D, 0, 0, MHA_D, 1, 1.0f);
  gemm_wmma_bf16<float, __bf16, true, 128><<<gProj, blk, 0, stream>>>(
      v, MHA_D, 0, 0, WvT, MHA_D, 0, 0, bv, Vb, MHA_D, 0, 0, MHA_D, 1, 1.0f);

  // 2) scores = (Q @ K^T) / 8 per (b,h): K_head rows are already B^T rows
  dim3 gScore(MHA_S / 128, MHA_S / BM, MHA_B * MHA_H);   // 16 x 16 x 32
  gemm_wmma_bf16<__bf16, float, true, 128><<<gScore, blk, 0, stream>>>(
      Qb, MHA_D, SD, MHA_DH,
      Kb, MHA_D, SD, MHA_DH,
      nullptr,
      attn, MHA_S, (long long)MHA_H * SS, SS,
      MHA_DH, MHA_H, 0.125f);

  // 3) softmax in place
  softmax_rows_2048<<<dim3((unsigned)(MHA_B * MHA_H * MHA_S)), blk, 0, stream>>>(attn);

  // 4) ctx = attn @ V per (b,h), BN=64 config, written into O[B,S,D]
  dim3 gCtx(1, MHA_S / BM, MHA_B * MHA_H);               // 1 x 16 x 32
  gemm_wmma_bf16<float, __bf16, false, 64><<<gCtx, blk, 0, stream>>>(
      attn, MHA_S, (long long)MHA_H * SS, SS,
      Vb, MHA_D, SD, MHA_DH,
      nullptr,
      Ob, MHA_D, SD, MHA_DH,
      MHA_S, MHA_H, 1.0f);

  // 5) output = O @ Wo^T(bf16) + bo
  gemm_wmma_bf16<__bf16, float, true, 128><<<gProj, blk, 0, stream>>>(
      Ob, MHA_D, 0, 0, WoT, MHA_D, 0, 0, bo, outp, MHA_D, 0, 0, MHA_D, 1, 1.0f);
}
